// MultiHeadAttention_16114717294921
// MI455X (gfx1250) — matrix operand order repaired and checked by reading, not compiled, not run on hardware
//
#include <hip/hip_runtime.h>

// Problem dims (fixed by the reference)
#define NB 8
#define NS 1024
#define NE 1024
#define NH 16
#define ND 64

typedef __attribute__((ext_vector_type(16))) __bf16 v16bf;
typedef __attribute__((ext_vector_type(8)))  float  v8f;

#if __has_builtin(__builtin_amdgcn_tensor_load_to_lds)
#define USE_TDM 1
#else
#define USE_TDM 0
#endif

// ---------------------------------------------------------------- helpers ---
static __device__ __forceinline__ unsigned short f2bf(float f) {
  // fp32 -> bf16 round-to-nearest-even
  unsigned int u = __builtin_bit_cast(unsigned int, f);
  u += 0x7FFFu + ((u >> 16) & 1u);
  return (unsigned short)(u >> 16);
}

// Load a 16-element bf16 fragment as two 16B chunks (ds_load_b128 / b128 global)
static __device__ __forceinline__ v16bf ld_frag2(const unsigned short* p0,
                                                 const unsigned short* p1) {
  union { uint4 q[2]; v16bf v; } u;
  u.q[0] = *(const uint4*)p0;
  u.q[1] = *(const uint4*)p1;
  return u.v;
}

static __device__ __forceinline__ v8f wmma_bf16(v16bf a, v16bf b, v8f c) {
  // D = A(16x32) * B(32x16) + C, fp32 accumulate
  return __builtin_amdgcn_wmma_f32_16x16x32_bf16(false, a, false, b, (short)0, c,
                                                 false, false);
}

// Fragment addressing per ISA 7.12.2 (bf16, wave32):
//   A (16x32): lane m=lane%16, half=lane/16 -> elems = row m, K runs
//              [8*half, 8*half+8) and [16+8*half, 16+8*half+8)   (contiguous!)
//   B (32x16): lane n=lane%16, half=lane/16 -> elems = col n, K run
//              [8*half, 8*half+8) and [16+8*half, 16+8*half+8)   (same as A)
//   C/D f32 (16x16): row = vgpr + 8*half, col = lane%16

#if USE_TDM
typedef __attribute__((ext_vector_type(4))) unsigned int u32x4;
typedef __attribute__((ext_vector_type(8))) int          i32x8;
typedef __attribute__((ext_vector_type(4))) int          i32x4;

// 2D TDM load: tile 64x64 bf16 -> LDS with pitch 72 bf16 (pad 4 dwords after
// every 32 dwords, i.e. after each 64-elem row). stride0 = elements between
// consecutive tile rows in memory.
static __device__ __forceinline__ void tdm_load_64x64(unsigned int lds_off,
                                                      const void* gptr,
                                                      unsigned int stride0) {
  unsigned long long ga = (unsigned long long)(uintptr_t)gptr;
  u32x4 g0;
  g0[0] = 1u;                                  // count = 1 valid descriptor
  g0[1] = lds_off;                             // LDS byte address
  g0[2] = (unsigned int)ga;                    // global_addr[31:0]
  g0[3] = (unsigned int)((ga >> 32) & 0x01FFFFFFull) | (2u << 30); // [56:32]|type=2
  i32x8 g1;
  g1[0] = (int)((1u << 16)      // data_size = 1 (2 bytes)
              | (1u << 20)      // pad_enable
              | (4u << 22)      // pad_interval: 32 dwords (one 64-elem row)
              | (3u << 25));    // pad_amount: 4 dwords (8 bf16 -> pitch 72)
  g1[1] = (int)((64u & 0xFFFFu) << 16);        // tensor_dim0[15:0] = 64
  g1[2] = (int)((64u & 0xFFFFu) << 16);        // dim0 hi=0 | tensor_dim1[15:0]=64
  g1[3] = (int)(64u << 16);                    // dim1 hi=0 | tile_dim0 = 64
  g1[4] = (int)(64u);                          // tile_dim1 = 64, tile_dim2 = 0
  g1[5] = (int)stride0;                        // tensor_dim0_stride[31:0]
  g1[6] = 0;                                   // stride0 hi | stride1 lo
  g1[7] = 0;                                   // stride1 hi
  i32x4 z4 = {0, 0, 0, 0};
  i32x8 z8 = {0, 0, 0, 0, 0, 0, 0, 0};
  // 6-arg variant (clang-23 / therock-10.0 headers)
  __builtin_amdgcn_tensor_load_to_lds(g0, g1, z4, z4, z8, 0);
}
#endif

// ------------------------------------------------- kernel 1: QKV projection ---
// Out = X[b] @ W[h] + bias[h];  TR=0: Out[b,h,s,d] bf16, TR=1: Out[b,h,d,s] bf16
// grid = (NS/64, NH, NB), block = 128 (4 waves); wave: 16 rows x 64 cols
template <bool TR>
__global__ void __launch_bounds__(128)
qkv_proj_kernel(const float* __restrict__ X,      // [B,S,E] fp32
                const float* __restrict__ W,      // [H,E,D] fp32
                const float* __restrict__ bias,   // [H,D]   fp32
                unsigned short* __restrict__ Out) // bf16
{
  __shared__ unsigned short a_lds[64][40]; // 64 rows x 32 K, row-major (80B pitch)
  __shared__ unsigned short bt_lds[64][40];// B transposed: 64 N x 32 K

  const int tid = threadIdx.x;
  const int wave = tid >> 5, lane = tid & 31;
  const int m16 = lane & 15, half = lane >> 4;
  const int row0 = blockIdx.x * 64;
  const int h = blockIdx.y, b = blockIdx.z;

  const float* A  = X + (size_t)b * NS * NE;
  const float* Wh = W + (size_t)h * NE * ND;

  v8f acc[4] = {};

  for (int k0 = 0; k0 < NE; k0 += 32) {
    // stage A tile 64x32 (fp32 -> bf16), float4 reads + b64 LDS writes
    for (int idx = tid; idx < 64 * 8; idx += 128) {
      int r = idx >> 3, c4 = (idx & 7) * 4;
      const float4 f = *(const float4*)&A[(size_t)(row0 + r) * NE + k0 + c4];
      ushort4 hq;
      hq.x = f2bf(f.x); hq.y = f2bf(f.y); hq.z = f2bf(f.z); hq.w = f2bf(f.w);
      *(ushort4*)&a_lds[r][c4] = hq;
    }
    // stage B tile 32x64 transposed -> bt_lds[n][k]
    for (int idx = tid; idx < 32 * 16; idx += 128) {
      int r = idx >> 4, c4 = (idx & 15) * 4;
      const float4 f = *(const float4*)&Wh[(size_t)(k0 + r) * ND + c4];
      bt_lds[c4 + 0][r] = f2bf(f.x);
      bt_lds[c4 + 1][r] = f2bf(f.y);
      bt_lds[c4 + 2][r] = f2bf(f.z);
      bt_lds[c4 + 3][r] = f2bf(f.w);
    }
    __syncthreads();

    const unsigned short* ar = &a_lds[wave * 16 + m16][0];
    v16bf af = ld_frag2(ar + 8 * half, ar + 16 + 8 * half);
#pragma unroll
    for (int t = 0; t < 4; t++) {
      const unsigned short* br = &bt_lds[t * 16 + m16][0];
      v16bf bf = ld_frag2(br + 8 * half, br + 16 + 8 * half);
      acc[t] = wmma_bf16(af, bf, acc[t]);
    }
    __syncthreads();
  }

#pragma unroll
  for (int t = 0; t < 4; t++)
#pragma unroll
    for (int r = 0; r < 8; r++) {
      int s = row0 + wave * 16 + r + 8 * half;
      int n = t * 16 + m16;
      unsigned short v = f2bf(acc[t][r] + bias[h * ND + n]);
      if (!TR)
        Out[(((size_t)b * NH + h) * NS + s) * ND + n] = v;
      else
        Out[(((size_t)b * NH + h) * ND + n) * NS + s] = v;
    }
}

// ------------------------------------------- kernel 2: flash attention core ---
// grid = (NS/64, NH, NB), block = 128 (4 waves); wave owns 16 q-rows, D=64 out
__global__ void __launch_bounds__(128)
attn_kernel(const unsigned short* __restrict__ Q,   // [B,H,S,D] bf16
            const unsigned short* __restrict__ K,   // [B,H,S,D] bf16
            const unsigned short* __restrict__ Vt,  // [B,H,D,S] bf16 (transposed)
            const float* __restrict__ pbias,        // [S,S] fp32
            const int* __restrict__ idx0,           // [S,S] i32
            const int* __restrict__ idx1,           // [S,S] i32
            const unsigned char* __restrict__ mask, // [S,S] bool
            unsigned short* __restrict__ O)         // [B,S,E] bf16 (head-joined)
{
  __shared__ unsigned short kbuf[64][72];   // [kv][d]
  __shared__ unsigned short vbufT[64][72];  // [d][kv]
  __shared__ unsigned short pbuf[4][16][72];// per-wave P scratch (D->A relayout)

  const int tid = threadIdx.x;
  const int wave = tid >> 5, lane = tid & 31;
  const int m16 = lane & 15, half = lane >> 4;
  const int q0 = blockIdx.x * 64;
  const int h = blockIdx.y, b = blockIdx.z;

  const size_t head = ((size_t)b * NH + h) * NS * ND;
  const unsigned short* Qh  = Q + head;
  const unsigned short* Kh  = K + head;
  const unsigned short* Vth = Vt + head; // [D][S]

  // Q fragments (A-operand): row is contiguous in d -> 2 x b128 per chunk
  const unsigned short* qrow = Qh + (size_t)(q0 + wave * 16 + m16) * ND;
  v16bf qf[2];
#pragma unroll
  for (int c = 0; c < 2; c++)
    qf[c] = ld_frag2(qrow + c * 32 + 8 * half, qrow + c * 32 + 16 + 8 * half);

  v8f oacc[4] = {};
  float mstat[8], lstat[8];
#pragma unroll
  for (int r = 0; r < 8; r++) { mstat[r] = -1e30f; lstat[r] = 0.0f; }
  const float scale = 0.125f; // 1/sqrt(64)

  for (int kv0 = 0; kv0 < NS; kv0 += 64) {
    if (kv0 > q0 + 63) break; // causal tail: fully-masked tiles contribute 0

    __syncthreads();
#if USE_TDM
    if (wave == 0) {
      tdm_load_64x64((unsigned int)(uintptr_t)&kbuf[0][0],
                     Kh + (size_t)kv0 * ND, ND);
      tdm_load_64x64((unsigned int)(uintptr_t)&vbufT[0][0],
                     Vth + kv0, NS);
      __builtin_amdgcn_s_wait_tensorcnt(0);
    }
#else
    for (int idx = tid; idx < 64 * 8; idx += 128) {
      int r = idx >> 3, c8 = (idx & 7) * 8;
      *(uint4*)&kbuf[r][c8]  = *(const uint4*)&Kh[(size_t)(kv0 + r) * ND + c8];
      *(uint4*)&vbufT[r][c8] = *(const uint4*)&Vth[(size_t)r * NS + kv0 + c8];
    }
#endif
    __syncthreads();

    // S = Q K^T : B-fragment cols are K rows -> contiguous in d
    float st[4][8];
#pragma unroll
    for (int t = 0; t < 4; t++) {
      v8f s = {};
      const unsigned short* kr = &kbuf[t * 16 + m16][0];
#pragma unroll
      for (int c = 0; c < 2; c++) {
        v16bf kf = ld_frag2(kr + c * 32 + 8 * half, kr + c * 32 + 16 + 8 * half);
        s = wmma_bf16(qf[c], kf, s);
      }
#pragma unroll
      for (int r = 0; r < 8; r++) st[t][r] = s[r];
    }

    // scale + position-bias gather + mask
#pragma unroll
    for (int t = 0; t < 4; t++)
#pragma unroll
      for (int r = 0; r < 8; r++) {
        int row = q0 + wave * 16 + r + 8 * half;
        int col = kv0 + t * 16 + m16;
        size_t ij = (size_t)row * NS + col;
        float sv = st[t][r] * scale + pbias[(size_t)idx0[ij] * NS + idx1[ij]];
        st[t][r] = mask[ij] ? -1e30f : sv;
      }

    // online softmax update (row = r + 8*half, cols across 16 lanes)
#pragma unroll
    for (int r = 0; r < 8; r++) {
      float v = fmaxf(fmaxf(st[0][r], st[1][r]), fmaxf(st[2][r], st[3][r]));
      v = fmaxf(v, __shfl_xor(v, 1, 32));
      v = fmaxf(v, __shfl_xor(v, 2, 32));
      v = fmaxf(v, __shfl_xor(v, 4, 32));
      v = fmaxf(v, __shfl_xor(v, 8, 32));
      float mnew = fmaxf(mstat[r], v);
      float alpha = __expf(mstat[r] - mnew);
      float rs = 0.0f;
#pragma unroll
      for (int t = 0; t < 4; t++) {
        float p = __expf(st[t][r] - mnew);
        st[t][r] = p;
        rs += p;
      }
      rs += __shfl_xor(rs, 1, 32);
      rs += __shfl_xor(rs, 2, 32);
      rs += __shfl_xor(rs, 4, 32);
      rs += __shfl_xor(rs, 8, 32);
      lstat[r] = lstat[r] * alpha + rs;
      mstat[r] = mnew;
#pragma unroll
      for (int t = 0; t < 4; t++) oacc[t][r] *= alpha;
    }

    // P: D-layout -> LDS -> A-layout fragments (wave-private; DS in-order/wave)
#pragma unroll
    for (int t = 0; t < 4; t++)
#pragma unroll
      for (int r = 0; r < 8; r++)
        pbuf[wave][r + 8 * half][t * 16 + m16] = f2bf(st[t][r]);

    const unsigned short* pr = &pbuf[wave][m16][0];
    v16bf pf[2];
#pragma unroll
    for (int c = 0; c < 2; c++)
      pf[c] = ld_frag2(pr + c * 32 + 8 * half, pr + c * 32 + 16 + 8 * half);

    // O += P V : B-fragment cols are V columns -> contiguous in vbufT rows
#pragma unroll
    for (int t = 0; t < 4; t++) {
      const unsigned short* vr = &vbufT[t * 16 + m16][0];
#pragma unroll
      for (int c = 0; c < 2; c++) {
        v16bf vf = ld_frag2(vr + c * 32 + 8 * half, vr + c * 32 + 16 + 8 * half);
        oacc[t] = wmma_bf16(pf[c], vf, oacc[t]);
      }
    }
  }

  // normalize and store head-joined O[b, s, h*D + d] as bf16
  unsigned short* Ob = O + ((size_t)b * NS + q0 + wave * 16) * NE + h * ND;
#pragma unroll
  for (int r = 0; r < 8; r++) {
    float inv = 1.0f / lstat[r];
#pragma unroll
    for (int t = 0; t < 4; t++)
      Ob[(size_t)(r + 8 * half) * NE + t * 16 + m16] = f2bf(oacc[t][r] * inv);
  }
}

// ----------------------------------------- kernel 3: output projection GEMM ---
// out[BS,E] (fp32) = O[BS,E] (bf16) @ Wo[E,E] (fp32) + bo
// grid = (BS/64, E/64), block = 128
__global__ void __launch_bounds__(128)
out_proj_kernel(const unsigned short* __restrict__ A, // [B*S, E] bf16
                const float* __restrict__ W,          // [E,E] fp32
                const float* __restrict__ bias,       // [E]
                float* __restrict__ C)                // [B*S, E] fp32
{
  __shared__ unsigned short a_lds[64][40];  // [m][k]
  __shared__ unsigned short bt_lds[64][40]; // transposed [n][k]

  const int tid = threadIdx.x;
  const int wave = tid >> 5, lane = tid & 31;
  const int m16 = lane & 15, half = lane >> 4;
  const int row0 = blockIdx.x * 64;
  const int col0 = blockIdx.y * 64;

  v8f acc[4] = {};

  for (int k0 = 0; k0 < NE; k0 += 32) {
    // stage A (already bf16): 64x32 via b128
    for (int idx = tid; idx < 64 * 4; idx += 128) {
      int r = idx >> 2, c8 = (idx & 3) * 8;
      *(uint4*)&a_lds[r][c8] = *(const uint4*)&A[(size_t)(row0 + r) * NE + k0 + c8];
    }
    // stage Wo tile 32x64 transposed (fp32 -> bf16)
    for (int idx = tid; idx < 32 * 16; idx += 128) {
      int r = idx >> 4, c4 = (idx & 15) * 4;
      const float4 f = *(const float4*)&W[(size_t)(k0 + r) * NE + col0 + c4];
      bt_lds[c4 + 0][r] = f2bf(f.x);
      bt_lds[c4 + 1][r] = f2bf(f.y);
      bt_lds[c4 + 2][r] = f2bf(f.z);
      bt_lds[c4 + 3][r] = f2bf(f.w);
    }
    __syncthreads();

    const unsigned short* ar = &a_lds[wave * 16 + m16][0];
    v16bf af = ld_frag2(ar + 8 * half, ar + 16 + 8 * half);
#pragma unroll
    for (int t = 0; t < 4; t++) {
      const unsigned short* br = &bt_lds[t * 16 + m16][0];
      v16bf bf = ld_frag2(br + 8 * half, br + 16 + 8 * half);
      acc[t] = wmma_bf16(af, bf, acc[t]);
    }
    __syncthreads();
  }

#pragma unroll
  for (int t = 0; t < 4; t++)
#pragma unroll
    for (int r = 0; r < 8; r++) {
      int m = row0 + wave * 16 + r + 8 * half;
      int n = col0 + t * 16 + m16;
      C[(size_t)m * NE + n] = acc[t][r] + bias[n];
    }
}

// -------------------------------------------------------------------- launch ---
extern "C" void kernel_launch(void* const* d_in, const int* in_sizes, int n_in,
                              void* d_out, int out_size, void* d_ws, size_t ws_size,
                              hipStream_t stream) {
  (void)in_sizes; (void)n_in; (void)out_size; (void)ws_size;

  const float* query = (const float*)d_in[0];
  const float* key   = (const float*)d_in[1];
  const float* value = (const float*)d_in[2];
  const float* Wq    = (const float*)d_in[3];
  const float* bq    = (const float*)d_in[4];
  const float* Wk    = (const float*)d_in[5];
  const float* bk    = (const float*)d_in[6];
  const float* Wv    = (const float*)d_in[7];
  const float* bv    = (const float*)d_in[8];
  const float* Wo    = (const float*)d_in[9];
  const float* bo    = (const float*)d_in[10];
  const float* pbias = (const float*)d_in[11];
  const int*   idx0  = (const int*)d_in[12];
  const int*   idx1  = (const int*)d_in[13];
  const unsigned char* mask = (const unsigned char*)d_in[14];

  // workspace: q,k [B,H,S,D] bf16, v [B,H,D,S] bf16, O [B,S,E] bf16 (16MB each)
  const size_t seg = (size_t)NB * NH * NS * ND * sizeof(unsigned short);
  unsigned short* qws = (unsigned short*)d_ws;
  unsigned short* kws = (unsigned short*)((char*)d_ws + seg);
  unsigned short* vws = (unsigned short*)((char*)d_ws + 2 * seg);
  unsigned short* ows = (unsigned short*)((char*)d_ws + 3 * seg);

  dim3 blk(128);
  dim3 gproj(NS / 64, NH, NB);
  qkv_proj_kernel<false><<<gproj, blk, 0, stream>>>(query, Wq, bq, qws);
  qkv_proj_kernel<false><<<gproj, blk, 0, stream>>>(key,   Wk, bk, kws);
  qkv_proj_kernel<true ><<<gproj, blk, 0, stream>>>(value, Wv, bv, vws);

  attn_kernel<<<gproj, blk, 0, stream>>>(qws, kws, vws, pbias, idx0, idx1, mask, ows);

  dim3 gout(NB * NS / 64, NE / 64);
  out_proj_kernel<<<gout, blk, 0, stream>>>(ows, Wo, bo, (float*)d_out);
}
